// MistralMoeLayer_17042430231376
// MI455X (gfx1250) — matrix operand order repaired and checked by reading, not compiled, not run on hardware
//
#include <hip/hip_runtime.h>
#include <stdint.h>

#define HDIM 2048
#define FDIM 8192
#define NEXP 8
#define NTOK 4096   // B*S = 2*2048

typedef __attribute__((ext_vector_type(16))) __bf16 v16bf;
typedef __attribute__((ext_vector_type(8)))  float  v8f;

union BfFragA { uint4 u4[2]; v16bf v; };
union BfFragB { unsigned u[8]; v16bf v; };

// pack two fp32 -> bf16 pair (truncation) in ONE v_perm_b32
static __device__ __forceinline__ unsigned pack_bf16x2(float lo, float hi) {
    return __builtin_amdgcn_perm(__float_as_uint(hi), __float_as_uint(lo), 0x07060302u);
}

// fp32 -> bf16 bits, round to nearest even (epilogue only)
static __device__ __forceinline__ unsigned short f2bf_bits(float f) {
    unsigned u = __float_as_uint(f);
    unsigned r = u + 0x7fffu + ((u >> 16) & 1u);
    return (unsigned short)(r >> 16);
}

static __device__ __forceinline__ void pack_b_frag(const float* __restrict__ p, size_t ld, BfFragB& B) {
    #pragma unroll
    for (int i = 0; i < 4; ++i) {
        B.u[i]     = pack_bf16x2(p[(size_t)(2*i) * ld], p[(size_t)(2*i+1) * ld]);
        B.u[4 + i] = pack_bf16x2(p[(size_t)(16+2*i) * ld], p[(size_t)(17+2*i) * ld]);
    }
}

// ---------------- zero output + routing counters ----------------
__global__ void zero_init_kernel(float* __restrict__ out, int n4, int* __restrict__ counts) {
    int i = blockIdx.x * blockDim.x + threadIdx.x;
    if (i < n4) reinterpret_cast<float4*>(out)[i] = make_float4(0.f, 0.f, 0.f, 0.f);
    if (i < NEXP) counts[i] = 0;
}

// ---------------- x fp32 -> bf16 ----------------
__global__ void cvt_x_bf16_kernel(const float* __restrict__ x, unsigned short* __restrict__ xb, int n4) {
    int i = blockIdx.x * blockDim.x + threadIdx.x;
    if (i >= n4) return;
    float4 f = reinterpret_cast<const float4*>(x)[i];
    uint2 o;
    o.x = pack_bf16x2(f.x, f.y);
    o.y = pack_bf16x2(f.z, f.w);
    reinterpret_cast<uint2*>(xb)[i] = o;
}

// ---------------- router: gate logits, top-2, softmax, compaction ----------------
__global__ void moe_router_kernel(const float* __restrict__ x, const float* __restrict__ gw,
                                  int* __restrict__ counts, int* __restrict__ toks,
                                  float* __restrict__ wts) {
    int t = blockIdx.x * blockDim.x + threadIdx.x;
    if (t >= NTOK) return;
    float lg[NEXP];
    #pragma unroll
    for (int e = 0; e < NEXP; ++e) lg[e] = 0.f;
    const float* xr = x + (size_t)t * HDIM;
    for (int h = 0; h < HDIM; ++h) {
        float xv = xr[h];
        const float* g = gw + (size_t)h * NEXP;
        #pragma unroll
        for (int e = 0; e < NEXP; ++e) lg[e] += xv * g[e];
    }
    // top-2, lowest index wins ties (jax.lax.top_k semantics)
    int e0 = 0; float v0 = lg[0];
    #pragma unroll
    for (int e = 1; e < NEXP; ++e) if (lg[e] > v0) { v0 = lg[e]; e0 = e; }
    int e1 = (e0 == 0) ? 1 : 0; float v1 = lg[e1];
    #pragma unroll
    for (int e = 0; e < NEXP; ++e) if (e != e0 && lg[e] > v1) { v1 = lg[e]; e1 = e; }
    // softmax over {v0, v1}, v0 >= v1
    float w1v = __expf(v1 - v0);
    float s   = 1.f + w1v;
    float w0  = 1.f / s;
    float w1n = w1v / s;
    int p0 = atomicAdd(&counts[e0], 1);
    toks[e0 * NTOK + p0] = t; wts[e0 * NTOK + p0] = w0;
    int p1 = atomicAdd(&counts[e1], 1);
    toks[e1 * NTOK + p1] = t; wts[e1 * NTOK + p1] = w1n;
}

// ---------------- GEMM1: h = silu(xg @ w1) * (xg @ w3), bf16 out ----------------
// Block: 8 waves, 64-row x 128-col tile. Wave: 16 cols, 4 M-subtiles (B reused 4x).
__global__ __launch_bounds__(256) void moe_gemm1_kernel(
    const unsigned short* __restrict__ xb,    // [NTOK, HDIM] bf16 bits
    const float* __restrict__ w1e,            // [HDIM, FDIM]
    const float* __restrict__ w3e,            // [HDIM, FDIM]
    const int*  __restrict__ toks,
    const int*  __restrict__ countp,
    unsigned short* __restrict__ hbuf)        // [NTOK, FDIM] bf16 bits
{
    const int cnt = *countp;
    const int mt  = blockIdx.x;                       // 64-row supertile
    if (mt * 64 >= cnt) return;
    const int lane = threadIdx.x & 31;
    const int mrow = lane & 15;
    const int hi   = lane >> 4;
    const int n0   = blockIdx.y * 128 + (threadIdx.x >> 5) * 16;

    const unsigned short* arow[4];
    #pragma unroll
    for (int s = 0; s < 4; ++s) {
        int p = mt * 64 + s * 16 + mrow;
        if (p >= cnt) p = cnt - 1;                    // clamp pad rows
        arow[s] = xb + (size_t)toks[p] * HDIM + hi * 8;
    }
    const float* b1p = w1e + (size_t)(hi * 8) * FDIM + n0 + mrow;
    const float* b3p = w3e + (size_t)(hi * 8) * FDIM + n0 + mrow;

    const v8f vz = {0.f,0.f,0.f,0.f,0.f,0.f,0.f,0.f};
    v8f acc1[4], acc3[4];
    #pragma unroll
    for (int s = 0; s < 4; ++s) { acc1[s] = vz; acc3[s] = vz; }

    for (int k0 = 0; k0 < HDIM; k0 += 32) {
        const float* r1 = b1p + (size_t)k0 * FDIM;
        const float* r3 = b3p + (size_t)k0 * FDIM;
        if (k0 + 64 < HDIM) {
            __builtin_prefetch(r1 + (size_t)64 * FDIM, 0, 1);
            __builtin_prefetch(r3 + (size_t)64 * FDIM, 0, 1);
        }
        BfFragB B1, B3;
        pack_b_frag(r1, FDIM, B1);
        pack_b_frag(r3, FDIM, B3);
        #pragma unroll
        for (int s = 0; s < 4; ++s) {
            BfFragA A;
            A.u4[0] = *reinterpret_cast<const uint4*>(arow[s] + k0);
            A.u4[1] = *reinterpret_cast<const uint4*>(arow[s] + k0 + 16);
            acc1[s] = __builtin_amdgcn_wmma_f32_16x16x32_bf16(false, A.v, false, B1.v,
                                                              (short)0, acc1[s], false, false);
            acc3[s] = __builtin_amdgcn_wmma_f32_16x16x32_bf16(false, A.v, false, B3.v,
                                                              (short)0, acc3[s], false, false);
        }
    }

    #pragma unroll
    for (int s = 0; s < 4; ++s) {
        #pragma unroll
        for (int r = 0; r < 8; ++r) {
            float g   = acc1[s][r];
            float sv  = g * __builtin_amdgcn_rcpf(1.f + __expf(-g));   // silu
            float val = sv * acc3[s][r];
            int pos = mt * 64 + s * 16 + hi * 8 + r;
            hbuf[(size_t)pos * FDIM + n0 + mrow] = f2bf_bits(val);
        }
    }
}

// ---------------- GEMM2: out[tok] += wt * (h @ w2) ----------------
// Block: 8 waves, 64-row x 256-col tile. Wave: 2 N-chunks x 4 M-subtiles
// (each A frag feeds 2 WMMAs; 8 A + 32 B loads per 8 WMMAs).
__global__ __launch_bounds__(256) void moe_gemm2_kernel(
    const unsigned short* __restrict__ hbuf,  // [NTOK, FDIM] bf16 bits
    const float* __restrict__ w2e,            // [FDIM, HDIM]
    const int*  __restrict__ toks,
    const float* __restrict__ wts,
    const int*  __restrict__ countp,
    float* __restrict__ out)                  // [NTOK, HDIM]
{
    const int cnt = *countp;
    const int mt  = blockIdx.x;
    if (mt * 64 >= cnt) return;
    const int lane = threadIdx.x & 31;
    const int mrow = lane & 15;
    const int hi   = lane >> 4;
    const int n0   = blockIdx.y * 256 + (threadIdx.x >> 5) * 32;   // 32 cols per wave

    const unsigned short* arow[4];
    #pragma unroll
    for (int s = 0; s < 4; ++s)
        arow[s] = hbuf + (size_t)(mt * 64 + s * 16 + mrow) * FDIM + hi * 8;
    const float* bp = w2e + (size_t)(hi * 8) * HDIM + n0 + mrow;

    const v8f vz = {0.f,0.f,0.f,0.f,0.f,0.f,0.f,0.f};
    v8f acc[4][2];
    #pragma unroll
    for (int s = 0; s < 4; ++s) { acc[s][0] = vz; acc[s][1] = vz; }

    for (int k0 = 0; k0 < FDIM; k0 += 32) {
        const float* r2 = bp + (size_t)k0 * HDIM;
        if (k0 + 64 < FDIM) __builtin_prefetch(r2 + (size_t)64 * HDIM, 0, 1);
        BfFragB B0, B1;
        pack_b_frag(r2,      HDIM, B0);
        pack_b_frag(r2 + 16, HDIM, B1);
        #pragma unroll
        for (int s = 0; s < 4; ++s) {
            BfFragA A;
            A.u4[0] = *reinterpret_cast<const uint4*>(arow[s] + k0);
            A.u4[1] = *reinterpret_cast<const uint4*>(arow[s] + k0 + 16);
            acc[s][0] = __builtin_amdgcn_wmma_f32_16x16x32_bf16(false, A.v, false, B0.v,
                                                                (short)0, acc[s][0], false, false);
            acc[s][1] = __builtin_amdgcn_wmma_f32_16x16x32_bf16(false, A.v, false, B1.v,
                                                                (short)0, acc[s][1], false, false);
        }
    }

    #pragma unroll
    for (int s = 0; s < 4; ++s) {
        int pos = mt * 64 + s * 16 + hi * 8;
        #pragma unroll
        for (int r = 0; r < 8; ++r) {
            if (pos + r < cnt) {
                int   tok = toks[pos + r];
                float tw  = wts[pos + r];
                float* orow = out + (size_t)tok * HDIM + n0 + mrow;
                atomicAdd(orow,      tw * acc[s][0][r]);
                atomicAdd(orow + 16, tw * acc[s][1][r]);
            }
        }
    }
}

extern "C" void kernel_launch(void* const* d_in, const int* in_sizes, int n_in,
                              void* d_out, int out_size, void* d_ws, size_t ws_size,
                              hipStream_t stream) {
    const float* x  = (const float*)d_in[0];   // [B,S,H]
    const float* gw = (const float*)d_in[1];   // [H,E]
    const float* w1 = (const float*)d_in[2];   // [E,H,F]
    const float* w3 = (const float*)d_in[3];   // [E,H,F]
    const float* w2 = (const float*)d_in[4];   // [E,F,H]
    float* out = (float*)d_out;

    char* ws = (char*)d_ws;
    int*   counts = (int*)ws;                 ws += 256;
    int*   toks   = (int*)ws;                 ws += (size_t)NEXP * NTOK * sizeof(int);
    float* wts    = (float*)ws;               ws += (size_t)NEXP * NTOK * sizeof(float);
    unsigned short* xb   = (unsigned short*)ws; ws += (size_t)NTOK * HDIM * 2;
    unsigned short* hbuf = (unsigned short*)ws; // NTOK*FDIM*2, reused per expert

    const int n4 = NTOK * HDIM / 4;
    zero_init_kernel<<<(n4 + 255) / 256, 256, 0, stream>>>(out, n4, counts);
    cvt_x_bf16_kernel<<<(n4 + 255) / 256, 256, 0, stream>>>(x, xb, n4);
    moe_router_kernel<<<(NTOK + 255) / 256, 256, 0, stream>>>(x, gw, counts, toks, wts);

    for (int e = 0; e < NEXP; ++e) {
        moe_gemm1_kernel<<<dim3(NTOK / 64, FDIM / 128), 256, 0, stream>>>(
            xb, w1 + (size_t)e * HDIM * FDIM, w3 + (size_t)e * HDIM * FDIM,
            toks + e * NTOK, counts + e, hbuf);
        moe_gemm2_kernel<<<dim3(NTOK / 64, HDIM / 256), 256, 0, stream>>>(
            hbuf, w2 + (size_t)e * FDIM * HDIM,
            toks + e * NTOK, wts + e * NTOK, counts + e, out);
    }
}
